// CrossAttention_90391881712384
// MI455X (gfx1250) — matrix operand order repaired and checked by reading, not compiled, not run on hardware
//
#include <hip/hip_runtime.h>

// ---------------------------------------------------------------------------
// Cross-attention for MI455X (gfx1250), wave32 + WMMA f16 (f32 accumulate).
//   prep  : fold 1x1-in convs into Q/K/V weights, fold softmax scale into Wq
//   qkv   : Q/K/V projections -> operand-native f16 "blobs" in workspace
//   flash : flash attention, d=256, double-buffered TDM-staged K/V in LDS
//   proj  : output 1x1 conv -> f32 [b][c][h*w]
// ---------------------------------------------------------------------------

typedef _Float16 f16x8  __attribute__((ext_vector_type(8)));
typedef _Float16 f16x16 __attribute__((ext_vector_type(16)));
typedef float    f32x8  __attribute__((ext_vector_type(8)));
typedef unsigned int u32x4 __attribute__((ext_vector_type(4)));
typedef int          i32x4 __attribute__((ext_vector_type(4)));
typedef int          i32x8 __attribute__((ext_vector_type(8)));

#define NBATCH 8
#define CIN    256
#define EMB    256
#define NTOK   4096

// workspace layout (bytes)
#define WQE_OFF 0x00000u            // f16 [256][256]  (scaled by 1/16)
#define WKE_OFF 0x20000u            // f16 [256][256]
#define WVE_OFF 0x40000u            // f16 [256][256]
#define WOH_OFF 0x60000u            // f16 [256][256]  (w_out converted)
#define BQE_OFF 0x80000u            // f32 [256]
#define BKE_OFF 0x80400u            // f32 [256]
#define BVE_OFF 0x80800u            // f32 [256]
#define QSW_OFF 0x90000ull                        // 16 MB : A-native blobs (qtile, dchunk)
#define KSW_OFF (QSW_OFF + (16ull << 20))         // 16 MB : B-native blobs (jtile, dchunk)
#define VSW_OFF (KSW_OFF + (16ull << 20))         // 16 MB : B-native blobs (jchunk, dtile)
#define OSW_OFF (VSW_OFF + (16ull << 20))         // 16 MB : B-native blobs (ntile, echunk)
#define BATCH_HALVES ((size_t)NTOK * EMB)         // 1M halves = 2MB per batch per tensor

// Tensor Data Mover availability (device pass only; 5-arg on clang-22/ROCm7.2,
// 6-arg on clang-23/amdgpu-toolchain — see CDNA5_HIP.md).
#if defined(__HIP_DEVICE_COMPILE__) && \
    __has_builtin(__builtin_amdgcn_tensor_load_to_lds) && \
    __has_builtin(__builtin_amdgcn_s_wait_tensorcnt)
#define HAVE_TDM 1
#else
#define HAVE_TDM 0
#endif

static __device__ __forceinline__ f32x8 wmma_f16(f16x16 a, f16x16 b, f32x8 c) {
  // (neg_a, A, neg_b, B, c_mod, C, reuse_a, reuse_b)
  return __builtin_amdgcn_wmma_f32_16x16x32_f16(false, a, false, b, (short)0, c,
                                                false, false);
}

union Frag { f16x16 v; f16x8 h[2]; _Float16 e[16]; };

#if HAVE_TDM
// One 16KB linear global->LDS copy via TDM (D# per CDNA5 ISA):
// 1-row tile, data_size=8B, tensor_dim0=tile_dim0=2048 elements.
static __device__ __forceinline__ void tdm_copy16k(unsigned lds_addr, const void* g) {
  const unsigned long long ga = (unsigned long long)(size_t)g;
  const u32x4 g0 = {
      1u,                                                    // count=1, user D#
      lds_addr,                                              // lds_addr
      (unsigned)(ga & 0xFFFFFFFFull),                        // global_addr[31:0]
      (unsigned)((ga >> 32) & 0x01FFFFFFull) | (2u << 30)    // addr[56:32] | type=2
  };
  const i32x8 g1 = {
      (int)(3u << 16),        // workgroup_mask=0, data_size=3 (8 bytes)
      (int)(0x0800u << 16),   // tensor_dim0[15:0] = 2048 at bits[63:48]
      (int)(1u << 16),        // tensor_dim0[31:16]=0 | tensor_dim1=1 at bits[95:80]
      (int)(0x0800u << 16),   // tile_dim0 = 2048 at bits[127:112]
      1,                      // tile_dim1 = 1
      2048,                   // tensor_dim0_stride[31:0]
      0, 0
  };
  const i32x4 z4 = {};
#if __clang_major__ >= 23
  const i32x8 z8 = {};
  __builtin_amdgcn_tensor_load_to_lds(g0, g1, z4, z4, z8, 0);
#else
  __builtin_amdgcn_tensor_load_to_lds(g0, g1, z4, z4, 0);
#endif
}
#endif

// ---------------------------------------------------------------------------
// Kernel 1: fold input 1x1 convs into Q/K/V weights; convert to f16.
// ---------------------------------------------------------------------------
__global__ __launch_bounds__(256) void prep_weights(
    const float* __restrict__ w_in1, const float* __restrict__ b_in1,
    const float* __restrict__ w_in2, const float* __restrict__ b_in2,
    const float* __restrict__ wq, const float* __restrict__ bq,
    const float* __restrict__ wk, const float* __restrict__ bk,
    const float* __restrict__ wv, const float* __restrict__ bv,
    const float* __restrict__ w_out, char* __restrict__ ws) {
  const int o = blockIdx.x;
  const int c = threadIdx.x;
  float aq = 0.f, ak = 0.f, av = 0.f;
  for (int e = 0; e < EMB; ++e) {
    const float q = wq[o * EMB + e], k = wk[o * EMB + e], v = wv[o * EMB + e];
    aq += q * w_in1[e * CIN + c];
    ak += k * w_in2[e * CIN + c];
    av += v * w_in2[e * CIN + c];
  }
  _Float16* WQE = (_Float16*)(ws + WQE_OFF);
  _Float16* WKE = (_Float16*)(ws + WKE_OFF);
  _Float16* WVE = (_Float16*)(ws + WVE_OFF);
  _Float16* WOH = (_Float16*)(ws + WOH_OFF);
  WQE[o * CIN + c] = (_Float16)(aq * 0.0625f);   // fold 256^-0.5 = 1/16
  WKE[o * CIN + c] = (_Float16)ak;
  WVE[o * CIN + c] = (_Float16)av;
  WOH[o * EMB + c] = (_Float16)w_out[o * EMB + c];
  if (c == 0) {
    float sq = 0.f, sk = 0.f, sv = 0.f;
    for (int e = 0; e < EMB; ++e) {
      sq += wq[o * EMB + e] * b_in1[e];
      sk += wk[o * EMB + e] * b_in2[e];
      sv += wv[o * EMB + e] * b_in2[e];
    }
    ((float*)(ws + BQE_OFF))[o] = (sq + bq[o]) * 0.0625f;
    ((float*)(ws + BKE_OFF))[o] = sk + bk[o];
    ((float*)(ws + BVE_OFF))[o] = sv + bv[o];
  }
}

// ---------------------------------------------------------------------------
// Kernel 2: Q/K/V projection -> operand-native 1KB blobs (lane-major, 32B/lane).
// Grid (ntile-groups=32, dgroups=8, batch=8), 256 threads = 8 waves.
// ---------------------------------------------------------------------------
__global__ __launch_bounds__(256) void qkv_kernel(const float* __restrict__ x,
                                                  const float* __restrict__ attn,
                                                  char* __restrict__ ws) {
  __shared__ __align__(16) _Float16 Qs[8][32][16];   // [wave][d-local][token]
  __shared__ __align__(16) _Float16 Ks[8][32][16];
  __shared__ __align__(16) _Float16 Vs[4][32][32];   // [pair][token-local][d-local]

  const int b = blockIdx.z;
  const int d0 = blockIdx.y * 32;
  const int w = threadIdx.x >> 5;
  const int lane = threadIdx.x & 31;
  const int laneLo = lane & 15;
  const int laneHi = lane >> 4;
  const int n0 = blockIdx.x * 128 + w * 16;

  const _Float16* WQE = (const _Float16*)(ws + WQE_OFF);
  const _Float16* WKE = (const _Float16*)(ws + WKE_OFF);
  const _Float16* WVE = (const _Float16*)(ws + WVE_OFF);
  const float* BQE = (const float*)(ws + BQE_OFF);
  const float* BKE = (const float*)(ws + BKE_OFF);
  const float* BVE = (const float*)(ws + BVE_OFF);

  const f32x8 zero = {};
  f32x8 dq[2], dk[2], dv[2];
#pragma unroll
  for (int t = 0; t < 2; ++t) { dq[t] = zero; dk[t] = zero; dv[t] = zero; }

  const int koffB = laneHi ? 8 : 0;    // B elements: [koffB..+7], [koffB+16..+23]
  const int koffA = laneHi ? 8 : 0;    // A elements: [koffA..+7], [koffA+16..+23]
  const float* xbase = x + (size_t)b * CIN * NTOK + n0 + laneLo;
  const float* abase = attn + (size_t)b * CIN * NTOK + n0 + laneLo;

  for (int c0 = 0; c0 < CIN; c0 += 32) {
    Frag bx, ba;
    const float* xp = xbase + (size_t)(c0 + koffB) * NTOK;
    const float* ap = abase + (size_t)(c0 + koffB) * NTOK;
#pragma unroll
    for (int i = 0; i < 16; ++i) {
      bx.e[i] = (_Float16)xp[(i + (i >= 8 ? 8 : 0)) * NTOK];
      ba.e[i] = (_Float16)ap[(i + (i >= 8 ? 8 : 0)) * NTOK];
    }
#pragma unroll
    for (int dt = 0; dt < 2; ++dt) {
      const int row = d0 + dt * 16 + laneLo;
      Frag a;
      const _Float16* p = WQE + row * CIN + c0 + koffA;
      a.h[0] = *(const f16x8*)p; a.h[1] = *(const f16x8*)(p + 16);
      dq[dt] = wmma_f16(a.v, bx.v, dq[dt]);
      p = WKE + row * CIN + c0 + koffA;
      a.h[0] = *(const f16x8*)p; a.h[1] = *(const f16x8*)(p + 16);
      dk[dt] = wmma_f16(a.v, ba.v, dk[dt]);
      p = WVE + row * CIN + c0 + koffA;
      a.h[0] = *(const f16x8*)p; a.h[1] = *(const f16x8*)(p + 16);
      dv[dt] = wmma_f16(a.v, ba.v, dv[dt]);
    }
  }

  // bias add + stage D tiles to LDS ([d-local][token] for Q/K; [tok][d] for V)
#pragma unroll
  for (int dt = 0; dt < 2; ++dt) {
#pragma unroll
    for (int j = 0; j < 8; ++j) {
      const int r = dt * 16 + j + 8 * laneHi;   // d-local 0..31
      const int gr = d0 + r;
      Qs[w][r][laneLo] = (_Float16)(dq[dt][j] + BQE[gr]);
      Ks[w][r][laneLo] = (_Float16)(dk[dt][j] + BKE[gr]);
      Vs[w >> 1][(w & 1) * 16 + laneLo][r] = (_Float16)(dv[dt][j] + BVE[gr]);
    }
  }
  __syncthreads();

  _Float16* QSW = (_Float16*)(ws + QSW_OFF) + (size_t)b * BATCH_HALVES;
  _Float16* KSW = (_Float16*)(ws + KSW_OFF) + (size_t)b * BATCH_HALVES;
  _Float16* VSW = (_Float16*)(ws + VSW_OFF) + (size_t)b * BATCH_HALVES;
  const int qtile = blockIdx.x * 8 + w;    // token tile 0..255
  const int dc = blockIdx.y;               // dchunk 0..7

  {  // Q blob: A-operand native (lane = token, K = d)
    Frag o;
#pragma unroll
    for (int i = 0; i < 16; ++i) {
      const int kk = i + (i >= 8 ? 8 : 0) + (laneHi ? 8 : 0);
      o.e[i] = Qs[w][kk][laneLo];
    }
    _Float16* dst = QSW + (size_t)(qtile * 8 + dc) * 512 + lane * 16;
    *(f16x8*)dst = o.h[0]; *(f16x8*)(dst + 8) = o.h[1];
  }
  {  // K blob: B-operand native (lane = key token, K = d)
    Frag o;
#pragma unroll
    for (int i = 0; i < 16; ++i) o.e[i] = Ks[w][i + (i >= 8 ? 8 : 0) + (laneHi ? 8 : 0)][laneLo];
    _Float16* dst = KSW + (size_t)(qtile * 8 + dc) * 512 + lane * 16;
    *(f16x8*)dst = o.h[0]; *(f16x8*)(dst + 8) = o.h[1];
  }
  {  // V blob: B-operand native (lane = d, K = key token); wave pair = 32 keys
    Frag o;
    const int p = w >> 1, q = w & 1;
#pragma unroll
    for (int i = 0; i < 16; ++i)
      o.e[i] = Vs[p][i + (i >= 8 ? 8 : 0) + (laneHi ? 8 : 0)][q * 16 + laneLo];
    const int jc = blockIdx.x * 4 + p;         // 32-key chunk 0..127
    const int dtg = blockIdx.y * 2 + q;        // 16-d tile 0..15
    _Float16* dst = VSW + (size_t)(jc * 16 + dtg) * 512 + lane * 16;
    *(f16x8*)dst = o.h[0]; *(f16x8*)(dst + 8) = o.h[1];
  }
}

// ---------------------------------------------------------------------------
// Kernel 3: flash attention. Grid (qgroups=64, batch=8), 128 threads = 4 waves;
// wave owns 16 queries, full d=256 f32 accumulator (128 VGPRs). K/V blobs for
// each 32-key chunk (16KB + 16KB, contiguous) are staged into LDS by the
// Tensor Data Mover with DOUBLE BUFFERING: the TDM fills buffer (jc+1)&1 while
// all waves compute from buffer jc&1. TENSORcnt retires in-order, so
// s_wait_tensorcnt(2) guarantees the current chunk's 2 copies have landed
// while the next chunk's 2 stay in flight.
// ---------------------------------------------------------------------------
__global__ __launch_bounds__(128) void flash_kernel(char* __restrict__ ws) {
#if HAVE_TDM
  __shared__ __align__(16) _Float16 KVst[2][16384]; // 2 x (K 16KB | V 16KB)
#else
  __shared__ __align__(16) _Float16 KVst[1][16384]; // epilogue scratch only
#endif
  __shared__ __align__(16) _Float16 Plds[4][512];   // 1KB/wave: P transpose scratch

  const int b = blockIdx.y;
  const int w = threadIdx.x >> 5;
  const int lane = threadIdx.x & 31;
  const int laneLo = lane & 15;
  const int laneHi = lane >> 4;
  const int qtile = blockIdx.x * 4 + w;   // 0..255

  const _Float16* QSW = (const _Float16*)(ws + QSW_OFF) + (size_t)b * BATCH_HALVES;
  const _Float16* KSW = (const _Float16*)(ws + KSW_OFF) + (size_t)b * BATCH_HALVES;
  const _Float16* VSW = (const _Float16*)(ws + VSW_OFF) + (size_t)b * BATCH_HALVES;
  _Float16* OSW = (_Float16*)(ws + OSW_OFF) + (size_t)b * BATCH_HALVES;

  const f32x8 zero = {};
  f32x8 O[16];
#pragma unroll
  for (int t = 0; t < 16; ++t) O[t] = zero;
  float m[8], l[8];
#pragma unroll
  for (int j = 0; j < 8; ++j) { m[j] = -1e30f; l[j] = 0.f; }

#if HAVE_TDM
  const unsigned lds0 = (unsigned)(size_t)(void*)&KVst[0][0];
  // prologue: stage chunk 0 into buffer 0 (2 TDM ops in flight)
  if (w == 0) {
    tdm_copy16k(lds0, (const void*)KSW);
    tdm_copy16k(lds0 + 16384u, (const void*)VSW);
  }
#endif

  for (int j0 = 0; j0 < NTOK; j0 += 32) {
    const int jt0 = j0 >> 4;
    const int jc = j0 >> 5;
    const int buf = jc & 1;
    (void)buf;
    if (j0 + 32 < NTOK) {  // prefetch next chunk's K/V blobs toward L2
      __builtin_prefetch(KSW + (size_t)((jt0 + 2) * 8) * 512 + lane * 16, 0, 0);
      __builtin_prefetch(VSW + (size_t)((jc + 1) * 16) * 512 + lane * 16, 0, 0);
    }

#if HAVE_TDM
    __syncthreads();               // all waves done reading buffer buf^1
    if (w == 0) {                  // wave-uniform: wave 0 drives the TDM
      if (j0 + 32 < NTOK) {
        const unsigned nb = lds0 + (unsigned)((buf ^ 1) * 32768);
        tdm_copy16k(nb, (const void*)(KSW + (size_t)((jt0 + 2) * 8) * 512));
        tdm_copy16k(nb + 16384u, (const void*)(VSW + (size_t)((jc + 1) * 16) * 512));
        __builtin_amdgcn_s_wait_tensorcnt(2);  // current chunk landed; next in flight
      } else {
        __builtin_amdgcn_s_wait_tensorcnt(0);
      }
    }
    __syncthreads();               // staged K/V for this chunk visible to all
#endif

    // ---- scores: S[16q x 32j] = Q (16x256) * K^T ----
    f32x8 s0 = zero, s1 = zero;
#pragma unroll
    for (int dcq = 0; dcq < 8; ++dcq) {
      Frag a, b0, b1;
      const f16x8* qa = (const f16x8*)(QSW + (size_t)(qtile * 8 + dcq) * 512 + lane * 16);
      a.h[0] = qa[0]; a.h[1] = qa[1];
#if HAVE_TDM
      const f16x8* k0 = (const f16x8*)&KVst[buf][(0 * 8 + dcq) * 512 + lane * 16];
      const f16x8* k1 = (const f16x8*)&KVst[buf][(1 * 8 + dcq) * 512 + lane * 16];
#else
      const f16x8* k0 = (const f16x8*)(KSW + (size_t)((jt0 + 0) * 8 + dcq) * 512 + lane * 16);
      const f16x8* k1 = (const f16x8*)(KSW + (size_t)((jt0 + 1) * 8 + dcq) * 512 + lane * 16);
#endif
      b0.h[0] = k0[0]; b0.h[1] = k0[1];
      b1.h[0] = k1[0]; b1.h[1] = k1[1];
      s0 = wmma_f16(a.v, b0.v, s0);
      s1 = wmma_f16(a.v, b1.v, s1);
    }

    // ---- online softmax (rows: VGPR idx + lane-half; cols: lanes) ----
    float al[8];
#pragma unroll
    for (int j = 0; j < 8; ++j) {
      float t = fmaxf(s0[j], s1[j]);
      t = fmaxf(t, __shfl_xor(t, 1, 32));
      t = fmaxf(t, __shfl_xor(t, 2, 32));
      t = fmaxf(t, __shfl_xor(t, 4, 32));
      t = fmaxf(t, __shfl_xor(t, 8, 32));
      const float mn = fmaxf(m[j], t);
      al[j] = __expf(m[j] - mn);
      m[j] = mn;
    }
#pragma unroll
    for (int j = 0; j < 8; ++j) {
      s0[j] = __expf(s0[j] - m[j]);
      s1[j] = __expf(s1[j] - m[j]);
    }
#pragma unroll
    for (int j = 0; j < 8; ++j) {
      float r = s0[j] + s1[j];
      r += __shfl_xor(r, 1, 32);
      r += __shfl_xor(r, 2, 32);
      r += __shfl_xor(r, 4, 32);
      r += __shfl_xor(r, 8, 32);
      l[j] = l[j] * al[j] + r;
    }
#pragma unroll
    for (int t = 0; t < 16; ++t)
#pragma unroll
      for (int j = 0; j < 8; ++j) O[t][j] *= al[j];

    // ---- P (f16) -> LDS in A-operand layout ----
#pragma unroll
    for (int tile = 0; tile < 2; ++tile)
#pragma unroll
      for (int j = 0; j < 8; ++j) {
        const int kk = laneLo + 16 * tile;           // key index 0..31
        const int mrow = j + 8 * laneHi;             // query row 0..15
        const int lp = mrow + (((kk >> 3) & 1) ? 16 : 0);
        const int ii = (kk & 7) + (kk >= 16 ? 8 : 0);
        Plds[w][lp * 16 + ii] = (_Float16)(tile ? s1[j] : s0[j]);
      }
    Frag ap;
    const f16x8* pp = (const f16x8*)&Plds[w][lane * 16];
    ap.h[0] = pp[0]; ap.h[1] = pp[1];

    // ---- O += P * V ----
#pragma unroll
    for (int dt = 0; dt < 16; ++dt) {
      Frag vb;
#if HAVE_TDM
      const f16x8* vp = (const f16x8*)&KVst[buf][8192 + dt * 512 + lane * 16];
#else
      const f16x8* vp = (const f16x8*)(VSW + (size_t)(jc * 16 + dt) * 512 + lane * 16);
#endif
      vb.h[0] = vp[0]; vb.h[1] = vp[1];
      O[dt] = wmma_f16(ap.v, vb.v, O[dt]);
    }
  }

  // ---- epilogue: normalize, transpose through LDS, emit O blobs ----
  __syncthreads();   // safe to reuse KVst[0] as per-wave scratch
  float inv[8];
#pragma unroll
  for (int j = 0; j < 8; ++j) inv[j] = 1.0f / l[j];
  _Float16* Ow = &KVst[0][w * 4096];   // [16 tokens][256 d] row-major, 8KB/wave
#pragma unroll
  for (int dt = 0; dt < 16; ++dt)
#pragma unroll
    for (int j = 0; j < 8; ++j) {
      const int mrow = j + 8 * laneHi;
      Ow[mrow * 256 + dt * 16 + laneLo] = (_Float16)(O[dt][j] * inv[j]);
    }
#pragma unroll
  for (int ec = 0; ec < 8; ++ec) {
    const _Float16* p = &Ow[laneLo * 256 + ec * 32 + (laneHi ? 16 : 0)];
    const f16x8 lo = *(const f16x8*)p;
    const f16x8 hi = *(const f16x8*)(p + 8);
    _Float16* dst = OSW + (size_t)(qtile * 8 + ec) * 512 + lane * 16;
    *(f16x8*)dst = lo; *(f16x8*)(dst + 8) = hi;
  }
}

// ---------------------------------------------------------------------------
// Kernel 4: output projection. Grid (ntile-groups=32, otiles=16, batch=8),
// 256 threads = 8 waves; wave = one 16-channel x 16-token output tile.
// ---------------------------------------------------------------------------
__global__ __launch_bounds__(256) void proj_kernel(const float* __restrict__ b_out,
                                                   const char* __restrict__ ws,
                                                   float* __restrict__ out) {
  const int b = blockIdx.z;
  const int o0 = blockIdx.y * 16;
  const int w = threadIdx.x >> 5;
  const int lane = threadIdx.x & 31;
  const int laneLo = lane & 15;
  const int laneHi = lane >> 4;
  const int ntile = blockIdx.x * 8 + w;
  const _Float16* WOH = (const _Float16*)(ws + WOH_OFF);
  const _Float16* OSW = (const _Float16*)(ws + OSW_OFF) + (size_t)b * BATCH_HALVES;
  const int koffA = laneHi ? 16 : 0;

  const f32x8 zero = {};
  f32x8 acc = zero;
#pragma unroll
  for (int ec = 0; ec < 8; ++ec) {
    Frag a, bb;
    const _Float16* p = WOH + (o0 + laneLo) * EMB + ec * 32 + koffA;
    a.h[0] = *(const f16x8*)p; a.h[1] = *(const f16x8*)(p + 8);
    const f16x8* q = (const f16x8*)(OSW + (size_t)(ntile * 8 + ec) * 512 + lane * 16);
    bb.h[0] = q[0]; bb.h[1] = q[1];
    acc = wmma_f16(a.v, bb.v, acc);
  }
#pragma unroll
  for (int j = 0; j < 8; ++j) {
    const int o = o0 + j + 8 * laneHi;
    out[((size_t)b * CIN + o) * NTOK + ntile * 16 + laneLo] = acc[j] + b_out[o];
  }
}

// ---------------------------------------------------------------------------
extern "C" void kernel_launch(void* const* d_in, const int* in_sizes, int n_in,
                              void* d_out, int out_size, void* d_ws, size_t ws_size,
                              hipStream_t stream) {
  (void)in_sizes; (void)n_in; (void)out_size; (void)ws_size;
  const float* x     = (const float*)d_in[0];
  const float* attn  = (const float*)d_in[1];
  const float* w_in1 = (const float*)d_in[2];
  const float* b_in1 = (const float*)d_in[3];
  const float* w_in2 = (const float*)d_in[4];
  const float* b_in2 = (const float*)d_in[5];
  const float* wq    = (const float*)d_in[6];
  const float* bq    = (const float*)d_in[7];
  const float* wk    = (const float*)d_in[8];
  const float* bk    = (const float*)d_in[9];
  const float* wv    = (const float*)d_in[10];
  const float* bv    = (const float*)d_in[11];
  const float* w_out = (const float*)d_in[12];
  const float* b_out = (const float*)d_in[13];
  char* ws = (char*)d_ws;
  float* out = (float*)d_out;

  prep_weights<<<dim3(256), dim3(256), 0, stream>>>(w_in1, b_in1, w_in2, b_in2,
                                                    wq, bq, wk, bk, wv, bv, w_out, ws);
  qkv_kernel<<<dim3(32, 8, NBATCH), dim3(256), 0, stream>>>(x, attn, ws);
  flash_kernel<<<dim3(64, NBATCH), dim3(128), 0, stream>>>(ws);
  proj_kernel<<<dim3(32, 16, NBATCH), dim3(256), 0, stream>>>(b_out, ws, out);
}
